// GroupedQueryAttention_1400159338706
// MI455X (gfx1250) — matrix operand order repaired and checked by reading, not compiled, not run on hardware
//
#include <hip/hip_runtime.h>
#include <hip/hip_bf16.h>

// Problem constants (match reference)
constexpr int kB   = 2;
constexpr int kS   = 4096;
constexpr int kH   = 256;
constexpr int kNH  = 8;
constexpr int kNKV = 4;
constexpr int kHD  = 32;
constexpr int kG   = kNH / kNKV;   // 2

typedef __attribute__((ext_vector_type(16))) __bf16 v16bf;
typedef __attribute__((ext_vector_type(8)))  float  v8f;
typedef __attribute__((ext_vector_type(4)))  unsigned int u32x4;
typedef __attribute__((ext_vector_type(4)))  int  i32x4;
typedef __attribute__((ext_vector_type(8)))  int  i32x8;

// A-fragment (16x32 bf16) per-lane K index for element e (0..15), lane half (0/1).
// ISA 7.12.2: lanes0-15 VGPR0..3 -> K=0..7, VGPR4..7 -> K=16..23; lanes16-31: +8.
__device__ __forceinline__ int a_kk(int e, int half) {
    int j = e >> 1, p = e & 1;
    return (j < 4) ? (2 * j + p + 8 * half) : (16 + 2 * (j - 4) + p + 8 * half);
}

// ---------------------------------------------------------------------------
// Tensor Data Mover: issue an async 2D tile load (bf16 elements) into LDS.
// D# packing per CDNA5 ISA:
//   group0: count=1 | lds_addr | global_addr[56:0] | type=2
//   group1: data_size=2B, tensor_dim0/1, tile_dim0/1, tensor_dim0_stride
// Tracked with TENSORcnt (s_wait_tensorcnt).
// ---------------------------------------------------------------------------
__device__ __forceinline__ void tdm_load_tile(
    unsigned lds_off, const void* gp,
    unsigned tile0, unsigned tile1,
    unsigned td0, unsigned td1, unsigned long long stride0_elems) {
    unsigned long long ga = (unsigned long long)(uintptr_t)gp;
    u32x4 g0;
    g0[0] = 1u;                                           // count=1 (valid user D#)
    g0[1] = lds_off;                                      // LDS byte address
    g0[2] = (unsigned)(ga & 0xffffffffu);                 // global_addr[31:0]
    g0[3] = (unsigned)((ga >> 32) & 0x01ffffffu)          // global_addr[56:32]
          | 0x80000000u;                                  // type=2 ("image")
    i32x8 g1;
    g1[0] = 0x00010000;                                   // data_size=1 -> 2 bytes
    g1[1] = (int)((td0 & 0xffffu) << 16);                 // tensor_dim0[15:0]
    g1[2] = (int)((td0 >> 16) | ((td1 & 0xffffu) << 16)); // td0[31:16] | td1[15:0]
    g1[3] = (int)((td1 >> 16) | (tile0 << 16));           // td1[31:16] | tile_dim0
    g1[4] = (int)(tile1 & 0xffffu);                       // tile_dim1 (tile_dim2=0)
    g1[5] = (int)(stride0_elems & 0xffffffffu);           // tensor_dim0_stride[31:0]
    g1[6] = (int)((stride0_elems >> 32) & 0xffffu);       // stride[47:32] (dim1 stride=0)
    g1[7] = 0;
    i32x4 z4 = {0, 0, 0, 0};
#if __clang_major__ >= 23
    i32x8 z8 = {0, 0, 0, 0, 0, 0, 0, 0};
    __builtin_amdgcn_tensor_load_to_lds(g0, g1, z4, z4, z8, 0);
#else
    __builtin_amdgcn_tensor_load_to_lds(g0, g1, z4, z4, 0);
#endif
}

// ---------------------------------------------------------------------------
// Kernel 1: QKV projection (bf16 WMMA, f32 accum) + RoPE + scale folding.
// One wave per (batch, 16-token tile, head-slot). Slots: 0..7 Q heads,
// 8..11 K heads, 12..15 V heads. Each wave produces a 16x32 output tile.
// Q -> Qb[b][h][t][d] bf16 (scaled by 1/sqrt(HD))
// K -> Ktb[b][kv][d][t] bf16 (pre-transposed for flash B-fragments)
// V -> Vb[b][kv][t][d] bf16
// ---------------------------------------------------------------------------
__global__ void __launch_bounds__(32) qkv_rope_kernel(
    const float* __restrict__ X,
    const float* __restrict__ Wq,
    const float* __restrict__ Wk,
    const float* __restrict__ Wv,
    __bf16* __restrict__ Qb,
    __bf16* __restrict__ Ktb,
    __bf16* __restrict__ Vb) {
    const int lane = threadIdx.x & 31;
    const int rowM = lane & 15;
    const int half = lane >> 4;
    const int dlo  = lane & 15;

    const int slot = blockIdx.x & 15;
    int tmp = blockIdx.x >> 4;
    const int st = tmp & (kS / 16 - 1);
    const int b  = tmp / (kS / 16);

    const float* Wp;
    int nbase;
    if (slot < 8)       { Wp = Wq; nbase = slot * kHD; }
    else if (slot < 12) { Wp = Wk; nbase = (slot - 8) * kHD; }
    else                { Wp = Wv; nbase = (slot - 12) * kHD; }

    const float* Xrow = X + ((size_t)(b * kS + st * 16 + rowM)) * kH;

    v8f c0 = {}, c1 = {};
    for (int k0 = 0; k0 < kH; k0 += 32) {
        v16bf a, b0, b1;
#pragma unroll
        for (int e = 0; e < 16; ++e) a[e] = (__bf16)Xrow[k0 + a_kk(e, half)];
        const float* wc = Wp + (size_t)k0;   // lane = N column of B-frag, elements = K
#pragma unroll
        for (int n = 0; n < 16; ++n) {
            b0[n] = (__bf16)wc[(size_t)(nbase + dlo) * kH + a_kk(n, half)];
            b1[n] = (__bf16)wc[(size_t)(nbase + dlo + 16) * kH + a_kk(n, half)];
        }
        c0 = __builtin_amdgcn_wmma_f32_16x16x32_bf16(false, a, false, b0, (short)0, c0, false, false);
        c1 = __builtin_amdgcn_wmma_f32_16x16x32_bf16(false, a, false, b1, (short)0, c1, false, false);
    }

    // RoPE (Q and K only). Pair (d, d+16) sits at identical (r, lane) of c0/c1.
    if (slot < 12) {
        const float invf = __expf(-(float)dlo * (9.210340371976184f / 16.0f)); // 10000^(-d/16)
#pragma unroll
        for (int r = 0; r < 8; ++r) {
            int t = st * 16 + r + 8 * half;
            float ang = (float)t * invf;
            float cs = __cosf(ang), sn = __sinf(ang);
            float x0 = c0[r], x1 = c1[r];
            c0[r] = x0 * cs - x1 * sn;
            c1[r] = x1 * cs + x0 * sn;
        }
    }

    if (slot < 8) {                               // Q, fold 1/sqrt(32)
        const float sc = 0.17677669529663687f;
        size_t base = ((size_t)(b * kNH + slot) * kS + st * 16) * kHD;
#pragma unroll
        for (int r = 0; r < 8; ++r) {
            int m = r + 8 * half;
            Qb[base + (size_t)m * kHD + dlo]      = (__bf16)(c0[r] * sc);
            Qb[base + (size_t)m * kHD + dlo + 16] = (__bf16)(c1[r] * sc);
        }
    } else if (slot < 12) {                       // K, transposed [d][t]
        int kv = slot - 8;
        size_t base = ((size_t)(b * kNKV + kv) * kHD) * kS + st * 16;
#pragma unroll
        for (int r = 0; r < 8; ++r) {
            int m = r + 8 * half;
            Ktb[base + (size_t)dlo * kS + m]        = (__bf16)c0[r];
            Ktb[base + (size_t)(dlo + 16) * kS + m] = (__bf16)c1[r];
        }
    } else {                                      // V
        int kv = slot - 12;
        size_t base = ((size_t)(b * kNKV + kv) * kS + st * 16) * kHD;
#pragma unroll
        for (int r = 0; r < 8; ++r) {
            int m = r + 8 * half;
            Vb[base + (size_t)m * kHD + dlo]      = (__bf16)c0[r];
            Vb[base + (size_t)m * kHD + dlo + 16] = (__bf16)c1[r];
        }
    }
}

// ---------------------------------------------------------------------------
// Kernel 2: causal flash attention, bf16 WMMA, f32 online softmax.
// Block = 4 waves; wave w owns 16 query rows. K/V tiles (32 keys) are
// streamed into per-wave LDS buffers by the Tensor Data Mover, double
// buffered: tile kt+1 is in flight while tile kt feeds the WMMAs.
// TENSORcnt synchronizes TDM completion (in-order per wave => wait<=2
// after issuing the next pair guarantees the current pair landed).
// ---------------------------------------------------------------------------
// LDS layout (per block, single allocation => base offset 0):
//   [0, 32768)       : 4 waves x 2 buffers x (K tile 2KB + V tile 2KB)
//   [32768, 36864)   : 4 waves x 1KB P-staging (16x32 bf16)
#define KV_BYTES_PER_WAVE 8192      // 2 buffers * 4096
#define P_BASE 32768

__global__ void __launch_bounds__(128) flash_kernel(
    const __bf16* __restrict__ Qb,
    const __bf16* __restrict__ Ktb,
    const __bf16* __restrict__ Vb,
    float* __restrict__ O) {
    __shared__ alignas(128) char smem[P_BASE + 4 * 1024];

    const int lane = threadIdx.x & 31;
    const int w    = threadIdx.x >> 5;
    const int rowM = lane & 15;
    const int half = lane >> 4;
    const int dlo  = lane & 15;

    const int qt = blockIdx.x % (kS / 64);
    int tmp = blockIdx.x / (kS / 64);
    const int h  = tmp % kNH;
    const int b  = tmp / kNH;
    const int kv = h / kG;               // repeat_interleave GQA mapping
    const int q0 = qt * 64 + w * 16;

    const __bf16* Kt_h = Ktb + (size_t)(b * kNKV + kv) * kHD * kS;
    const __bf16* V_h  = Vb  + (size_t)(b * kNKV + kv) * kS * kHD;

    // Q fragment (A layout), 1/sqrt(HD) already folded in.
    const __bf16* Qrow = Qb + ((size_t)((b * kNH + h) * kS) + q0 + rowM) * kHD;
    v16bf qa;
#pragma unroll
    for (int e = 0; e < 16; ++e) qa[e] = Qrow[a_kk(e, half)];

    v8f acc0 = {}, acc1 = {};
    float mrow[8], lrow[8];
#pragma unroll
    for (int r = 0; r < 8; ++r) { mrow[r] = -1e30f; lrow[r] = 0.0f; }

    __bf16* Pw = (__bf16*)(smem + P_BASE + w * 1024);
    const unsigned kvbase = (unsigned)(w * KV_BYTES_PER_WAVE);

    // Issue TDM loads for one 32-key tile into buffer `buf`.
    auto issue_tile = [&](int kt, int buf) {
        unsigned off = kvbase + (unsigned)buf * 4096u;
        // K tile: 32 rows (d), 32 keys each, row stride = S elements (bf16).
        tdm_load_tile(off, Kt_h + kt * 32,
                      /*tile0=*/32, /*tile1=*/32,
                      /*td0=*/(unsigned)kS, /*td1=*/kHD,
                      /*stride0=*/(unsigned long long)kS);
        // V tile: 32 keys x 32 dims, fully contiguous 2KB -> 1D transfer.
        tdm_load_tile(off + 2048u, V_h + (size_t)kt * 32 * kHD,
                      /*tile0=*/1024, /*tile1=*/0,
                      /*td0=*/0x7fff0000u, /*td1=*/1,
                      /*stride0=*/1024ull);
    };

    const int ntiles = (qt * 64 + 64) / 32;   // identical for all 4 waves
    issue_tile(0, 0);

    for (int kt = 0; kt < ntiles; ++kt) {
        const int kbase = kt * 32;
        const int buf = kt & 1;

        if (kt + 1 < ntiles) {
            // Previous reads of buf^1 have retired (results already consumed),
            // make that explicit, then prefetch the next tile behind compute.
            asm volatile("s_wait_dscnt 0x0" ::: "memory");
            issue_tile(kt + 1, buf ^ 1);
            __builtin_amdgcn_s_wait_tensorcnt((short)2);  // current tile landed
        } else {
            __builtin_amdgcn_s_wait_tensorcnt((short)0);
        }

        const char* kvp = smem + kvbase + (unsigned)buf * 4096u;
        // K^T B-fragments from LDS: lane = key column, elements = d rows.
        const __bf16* ktl = (const __bf16*)kvp;
        v16bf kb0, kb1;
#pragma unroll
        for (int e = 0; e < 16; ++e) { kb0[e] = ktl[a_kk(e, half) * 32 + dlo]; kb1[e] = ktl[a_kk(e, half) * 32 + dlo + 16]; }

        v8f z = {};
        v8f s0 = __builtin_amdgcn_wmma_f32_16x16x32_bf16(false, qa, false, kb0, (short)0, z, false, false);
        v8f s1 = __builtin_amdgcn_wmma_f32_16x16x32_bf16(false, qa, false, kb1, (short)0, z, false, false);

#pragma unroll
        for (int r = 0; r < 8; ++r) {
            int qi = q0 + r + 8 * half;
            int j0 = kbase + dlo, j1 = j0 + 16;
            float e0 = (j0 <= qi) ? s0[r] : -1e30f;
            float e1 = (j1 <= qi) ? s1[r] : -1e30f;

            float mx = fmaxf(e0, e1);
#pragma unroll
            for (int off = 8; off >= 1; off >>= 1)
                mx = fmaxf(mx, __shfl_xor(mx, off, 16));
            float mnew = fmaxf(mrow[r], mx);
            float corr = __expf(mrow[r] - mnew);
            float p0 = __expf(e0 - mnew);
            float p1 = __expf(e1 - mnew);
            float ps = p0 + p1;
#pragma unroll
            for (int off = 8; off >= 1; off >>= 1)
                ps += __shfl_xor(ps, off, 16);
            lrow[r] = lrow[r] * corr + ps;
            mrow[r] = mnew;
            acc0[r] *= corr;
            acc1[r] *= corr;

            Pw[(r + 8 * half) * 32 + dlo]      = (__bf16)p0;
            Pw[(r + 8 * half) * 32 + dlo + 16] = (__bf16)p1;
        }
        __syncthreads();

        // Repack P (D layout) -> A fragment through LDS.
        v16bf pa;
#pragma unroll
        for (int e = 0; e < 16; ++e) pa[e] = Pw[rowM * 32 + a_kk(e, half)];

        // V B-fragments from LDS: lane = dim column, elements = key rows.
        const __bf16* vl = (const __bf16*)(kvp + 2048);
        v16bf vb0, vb1;
#pragma unroll
        for (int e = 0; e < 16; ++e) { vb0[e] = vl[a_kk(e, half) * 32 + dlo]; vb1[e] = vl[a_kk(e, half) * 32 + dlo + 16]; }

        acc0 = __builtin_amdgcn_wmma_f32_16x16x32_bf16(false, pa, false, vb0, (short)0, acc0, false, false);
        acc1 = __builtin_amdgcn_wmma_f32_16x16x32_bf16(false, pa, false, vb1, (short)0, acc1, false, false);
        __syncthreads();
    }

    // Normalize and write O[b][t][h*32 + d] (f32).
#pragma unroll
    for (int r = 0; r < 8; ++r) {
        int t = q0 + r + 8 * half;
        float inv = 1.0f / lrow[r];
        size_t o = ((size_t)(b * kS + t)) * kH + h * kHD;
        O[o + dlo]      = acc0[r] * inv;
        O[o + dlo + 16] = acc1[r] * inv;
    }
}

// ---------------------------------------------------------------------------
// Kernel 3: output projection Y = O @ Wo^T, bf16 WMMA, f32 out.
// One wave per 16x16 output tile over [B*S, H].
// ---------------------------------------------------------------------------
__global__ void __launch_bounds__(32) outproj_kernel(
    const float* __restrict__ O,
    const float* __restrict__ Wo,
    float* __restrict__ Y) {
    const int lane = threadIdx.x & 31;
    const int rowM = lane & 15;
    const int half = lane >> 4;

    const int nt = blockIdx.x & (kH / 16 - 1);
    const int mt = blockIdx.x >> 4;
    const int nbase = nt * 16;

    const float* Orow = O + ((size_t)(mt * 16 + rowM)) * kH;
    v8f c = {};
    for (int k0 = 0; k0 < kH; k0 += 32) {
        v16bf a, bf;
#pragma unroll
        for (int e = 0; e < 16; ++e) a[e] = (__bf16)Orow[k0 + a_kk(e, half)];
        const float* wc = Wo + (size_t)k0;
#pragma unroll
        for (int n = 0; n < 16; ++n) bf[n] = (__bf16)wc[(size_t)(nbase + rowM) * kH + a_kk(n, half)];
        c = __builtin_amdgcn_wmma_f32_16x16x32_bf16(false, a, false, bf, (short)0, c, false, false);
    }
#pragma unroll
    for (int r = 0; r < 8; ++r) {
        int m = mt * 16 + r + 8 * half;
        Y[(size_t)m * kH + nbase + (lane & 15)] = c[r];
    }
}

// ---------------------------------------------------------------------------
extern "C" void kernel_launch(void* const* d_in, const int* in_sizes, int n_in,
                              void* d_out, int out_size, void* d_ws, size_t ws_size,
                              hipStream_t stream) {
    (void)in_sizes; (void)n_in; (void)out_size; (void)ws_size;
    const float* X  = (const float*)d_in[0];
    const float* Wq = (const float*)d_in[1];
    const float* Wk = (const float*)d_in[2];
    const float* Wv = (const float*)d_in[3];
    const float* Wo = (const float*)d_in[4];

    char* ws = (char*)d_ws;
    const size_t qb_bytes = (size_t)kB * kNH  * kS * kHD * sizeof(__bf16);  // 4 MB
    const size_t kb_bytes = (size_t)kB * kNKV * kHD * kS * sizeof(__bf16);  // 2 MB
    const size_t vb_bytes = (size_t)kB * kNKV * kS * kHD * sizeof(__bf16);  // 2 MB
    __bf16* Qb  = (__bf16*)ws;
    __bf16* Ktb = (__bf16*)(ws + qb_bytes);
    __bf16* Vb  = (__bf16*)(ws + qb_bytes + kb_bytes);
    float*  O   = (float*)(ws + qb_bytes + kb_bytes + vb_bytes);            // 8 MB

    qkv_rope_kernel<<<kB * (kS / 16) * 16, 32, 0, stream>>>(X, Wq, Wk, Wv, Qb, Ktb, Vb);
    flash_kernel<<<kB * kNH * (kS / 64), 128, 0, stream>>>(Qb, Ktb, Vb, O);
    outproj_kernel<<<(kB * kS / 16) * (kH / 16), 32, 0, stream>>>(O, Wo, (float*)d_out);
}
